// MultiHeadAttention_73864847557011
// MI455X (gfx1250) — matrix operand order repaired and checked by reading, not compiled, not run on hardware
//
#include <hip/hip_runtime.h>
#include <hip/hip_bf16.h>
#include <stdint.h>

// ---------------------------------------------------------------------------
// MI455X (gfx1250) sliding-window attention, bf16 WMMA path.
//   B=2, S=4096, D=E=1024, H=16, hd=64, w=256. fp32 in/out, bf16 matmul
//   inputs, fp32 accumulation + fp32 softmax.
// ---------------------------------------------------------------------------

typedef __attribute__((ext_vector_type(16))) __bf16 v16bf;
typedef __attribute__((ext_vector_type(8)))  __bf16 bf16x8;
typedef __attribute__((ext_vector_type(4)))  __bf16 bf16x4;
typedef __attribute__((ext_vector_type(8)))  float  v8f;

#define WMMA_BF16(A, Bm, C) \
  __builtin_amdgcn_wmma_f32_16x16x32_bf16(false, (A), false, (Bm), (short)0, (C), false, false)

__device__ __forceinline__ v8f zerov8() {
  v8f z = {0.f, 0.f, 0.f, 0.f, 0.f, 0.f, 0.f, 0.f};
  return z;
}

// A-matrix 16x32 bf16 fragment (interleaved layout):
//   element e (0..7)  -> K = h8 + e        (h8 = (lane>=16)?8:0)
//   element e (8..15) -> K = 16 + h8 + (e-8)
__device__ __forceinline__ v16bf load_a_frag(const __bf16* row, int h8) {
  bf16x4 p0 = *(const bf16x4*)(row + h8);
  bf16x4 p1 = *(const bf16x4*)(row + h8 + 4);
  bf16x4 p2 = *(const bf16x4*)(row + 16 + h8);
  bf16x4 p3 = *(const bf16x4*)(row + 16 + h8 + 4);
  v16bf r;
  r[0] = p0[0]; r[1] = p0[1]; r[2]  = p0[2]; r[3]  = p0[3];
  r[4] = p1[0]; r[5] = p1[1]; r[6]  = p1[2]; r[7]  = p1[3];
  r[8] = p2[0]; r[9] = p2[1]; r[10] = p2[2]; r[11] = p2[3];
  r[12] = p3[0]; r[13] = p3[1]; r[14] = p3[2]; r[15] = p3[3];
  return r;
}

// B-matrix 32x16 bf16 fragment (interleaved layout, same as A):
//   element e (0..7)  -> K = h16/2 + e, element e (8..15) -> K = 16 + h16/2 + (e-8)
//   (h16 = (lane>=16)?16:0)
__device__ __forceinline__ v16bf load_b_frag(const __bf16* row, int h16) {
  bf16x8 p0 = *(const bf16x8*)(row + (h16 >> 1));
  bf16x8 p1 = *(const bf16x8*)(row + 16 + (h16 >> 1));
  v16bf r;
  r[0] = p0[0]; r[1] = p0[1]; r[2]  = p0[2]; r[3]  = p0[3];
  r[4] = p0[4]; r[5] = p0[5]; r[6]  = p0[6]; r[7]  = p0[7];
  r[8] = p1[0]; r[9] = p1[1]; r[10] = p1[2]; r[11] = p1[3];
  r[12] = p1[4]; r[13] = p1[5]; r[14] = p1[6]; r[15] = p1[7];
  return r;
}

// ---------------------------------------------------------------------------
// Kernel 1: qkv = x @ Wqkv^T + bqkv, scattered into q/k/v [B,H,S,64].
// M=8192 (B*S), N=3072, K=1024. Block tile 128x128, K-chunk 32.
// ---------------------------------------------------------------------------
__global__ __launch_bounds__(256) void k_qkv_gemm(
    const float* __restrict__ x, const float* __restrict__ W,
    const float* __restrict__ bias, float* __restrict__ qo,
    float* __restrict__ ko, float* __restrict__ vo) {
  __shared__ __bf16 As[128][40];  // 32 k + 8 pad (80B rows: aligned, bank-clean)
  __shared__ __bf16 Bs[128][40];

  const int tid  = threadIdx.x;
  const int lane = tid & 31, wid = tid >> 5;
  const int l15 = lane & 15, hf = lane >> 4;
  const int h8 = hf << 3, h16 = hf << 4;
  const int mbase = blockIdx.y * 128;
  const int nbase = blockIdx.x * 128;
  const int wr = wid >> 1, wc = wid & 1;  // 4x2 wave grid, wave tile 32x64

  v8f acc[2][4];
#pragma unroll
  for (int mt = 0; mt < 2; ++mt)
#pragma unroll
    for (int nt = 0; nt < 4; ++nt) acc[mt][nt] = zerov8();

  const int r     = tid >> 1;
  const int halfk = (tid & 1) << 4;
  const float* aptr = x + (size_t)(mbase + r) * 1024 + halfk;
  const float* bptr = W + (size_t)(nbase + r) * 1024 + halfk;

  for (int kc = 0; kc < 1024; kc += 32) {
    if (kc + 32 < 1024) {  // pull next chunk toward L2/L0 (global_prefetch_b8)
      __builtin_prefetch(aptr + kc + 32, 0, 1);
      __builtin_prefetch(bptr + kc + 32, 0, 1);
    }
    const float4* a4 = (const float4*)(aptr + kc);
    const float4* b4 = (const float4*)(bptr + kc);
#pragma unroll
    for (int u = 0; u < 4; ++u) {
      float4 fa = a4[u];
      float4 fb = b4[u];
      As[r][halfk + u * 4 + 0] = (__bf16)fa.x;
      As[r][halfk + u * 4 + 1] = (__bf16)fa.y;
      As[r][halfk + u * 4 + 2] = (__bf16)fa.z;
      As[r][halfk + u * 4 + 3] = (__bf16)fa.w;
      Bs[r][halfk + u * 4 + 0] = (__bf16)fb.x;
      Bs[r][halfk + u * 4 + 1] = (__bf16)fb.y;
      Bs[r][halfk + u * 4 + 2] = (__bf16)fb.z;
      Bs[r][halfk + u * 4 + 3] = (__bf16)fb.w;
    }
    __syncthreads();

    v16bf af[2], bv[4];
#pragma unroll
    for (int mt = 0; mt < 2; ++mt)
      af[mt] = load_a_frag(&As[wr * 32 + mt * 16 + l15][0], h8);
#pragma unroll
    for (int nt = 0; nt < 4; ++nt)
      bv[nt] = load_b_frag(&Bs[wc * 64 + nt * 16 + l15][0], h16);
#pragma unroll
    for (int mt = 0; mt < 2; ++mt)
#pragma unroll
      for (int nt = 0; nt < 4; ++nt)
        acc[mt][nt] = WMMA_BF16(af[mt], bv[nt], acc[mt][nt]);
    __syncthreads();
  }

  // Epilogue: +bias, scatter into q/k/v [B,H,S,64].
#pragma unroll
  for (int mt = 0; mt < 2; ++mt)
#pragma unroll
    for (int nt = 0; nt < 4; ++nt) {
      int nglob = nbase + wc * 64 + nt * 16 + l15;
      int sel = nglob >> 10, e = nglob & 1023;
      int hh = e >> 6, dd = e & 63;
      float bval = bias[nglob];
      float* op = (sel == 0) ? qo : ((sel == 1) ? ko : vo);
#pragma unroll
      for (int i = 0; i < 8; ++i) {
        int mrow = mbase + wr * 32 + mt * 16 + h8 + i;  // C layout: row = i + h8
        int bb = mrow >> 12, ss = mrow & 4095;
        op[(((size_t)(bb * 16 + hh) * 4096 + ss) << 6) + dd] = acc[mt][nt][i] + bval;
      }
    }
}

// ---------------------------------------------------------------------------
// Kernel 2: flash-style banded attention.
// Grid (S/128, H, B); 8 waves, each owns 16 query rows. 20 iters x 32 keys
// cover the workgroup's union window [q0-256, q0+384).
// ---------------------------------------------------------------------------
__global__ __launch_bounds__(256) void k_attn(
    const float* __restrict__ q, const float* __restrict__ k,
    const float* __restrict__ v, const int* __restrict__ pmask,
    float* __restrict__ ctx) {
  __shared__ __bf16 Ks[32][72];      // 32 keys x 64 d (+8 pad)
  __shared__ __bf16 Vt[64][40];      // V transposed: 64 d x 32 keys (+8 pad)
  __shared__ __bf16 Ps[8][16][40];   // per-wave P tile: 16 q x 32 keys (+8 pad)

  const int tid  = threadIdx.x;
  const int lane = tid & 31, wv = tid >> 5;
  const int l15 = lane & 15, hf = lane >> 4;
  const int h8 = hf << 3, h16 = hf << 4;
  const int q0   = blockIdx.x * 128;
  const int head = blockIdx.y;
  const int bb   = blockIdx.z;

  const size_t base = ((size_t)(bb * 16 + head)) * 4096 * 64;
  const float* qh = q + base;
  const float* kh = k + base;
  const float* vh = v + base;
  const int*   pm = pmask + bb * 4096;

  // Preload this wave's 16x64 q tile as 2 A-fragments (scaled by 1/sqrt(64)).
  v16bf qf[2];
  {
    const int qi = q0 + wv * 16 + l15;
    const float* qr = qh + (size_t)qi * 64;
#pragma unroll
    for (int c = 0; c < 2; ++c) {
      const float4* p  = (const float4*)(qr + c * 32 + h8);
      const float4* p2 = (const float4*)(qr + c * 32 + 16 + h8);
      float4 f0 = p[0], f1 = p[1], f2 = p2[0], f3 = p2[1];
      v16bf a;
      a[0]  = (__bf16)(f0.x * 0.125f); a[1]  = (__bf16)(f0.y * 0.125f);
      a[2]  = (__bf16)(f0.z * 0.125f); a[3]  = (__bf16)(f0.w * 0.125f);
      a[4]  = (__bf16)(f1.x * 0.125f); a[5]  = (__bf16)(f1.y * 0.125f);
      a[6]  = (__bf16)(f1.z * 0.125f); a[7]  = (__bf16)(f1.w * 0.125f);
      a[8]  = (__bf16)(f2.x * 0.125f); a[9]  = (__bf16)(f2.y * 0.125f);
      a[10] = (__bf16)(f2.z * 0.125f); a[11] = (__bf16)(f2.w * 0.125f);
      a[12] = (__bf16)(f3.x * 0.125f); a[13] = (__bf16)(f3.y * 0.125f);
      a[14] = (__bf16)(f3.z * 0.125f); a[15] = (__bf16)(f3.w * 0.125f);
      qf[c] = a;
    }
  }

  v8f o[4];
#pragma unroll
  for (int t = 0; t < 4; ++t) o[t] = zerov8();
  float mrun[8], lrun[8];
#pragma unroll
  for (int i = 0; i < 8; ++i) { mrun[i] = -1e30f; lrun[i] = 0.f; }

  const int sr = tid >> 3;         // staging key row 0..31
  const int c8 = (tid & 7) << 3;   // staging d-column base

  for (int it = 0; it < 20; ++it) {
    const int kb = q0 - 256 + it * 32;

    // Stage K (row-major) and V (transposed) for 32 keys, fp32 -> bf16.
    {
      int key = kb + sr;
      bool inr = (key >= 0) && (key < 4096);
      size_t off = (size_t)(inr ? key : 0) * 64 + c8;
      float4 z = {0.f, 0.f, 0.f, 0.f};
      const float4* kp = (const float4*)(kh + off);
      const float4* vp = (const float4*)(vh + off);
      float4 ka = inr ? kp[0] : z, kc = inr ? kp[1] : z;
      float4 va = inr ? vp[0] : z, vc = inr ? vp[1] : z;
      Ks[sr][c8 + 0] = (__bf16)ka.x; Ks[sr][c8 + 1] = (__bf16)ka.y;
      Ks[sr][c8 + 2] = (__bf16)ka.z; Ks[sr][c8 + 3] = (__bf16)ka.w;
      Ks[sr][c8 + 4] = (__bf16)kc.x; Ks[sr][c8 + 5] = (__bf16)kc.y;
      Ks[sr][c8 + 6] = (__bf16)kc.z; Ks[sr][c8 + 7] = (__bf16)kc.w;
      Vt[c8 + 0][sr] = (__bf16)va.x; Vt[c8 + 1][sr] = (__bf16)va.y;
      Vt[c8 + 2][sr] = (__bf16)va.z; Vt[c8 + 3][sr] = (__bf16)va.w;
      Vt[c8 + 4][sr] = (__bf16)vc.x; Vt[c8 + 5][sr] = (__bf16)vc.y;
      Vt[c8 + 6][sr] = (__bf16)vc.z; Vt[c8 + 7][sr] = (__bf16)vc.w;
    }
    __syncthreads();

    // Scores: two 16x16 tiles (keys kb..kb+15 and kb+16..kb+31), K=64.
    v8f s0 = zerov8(), s1 = zerov8();
#pragma unroll
    for (int c = 0; c < 2; ++c) {
      v16bf bk0 = load_b_frag(&Ks[l15][c * 32], h16);
      v16bf bk1 = load_b_frag(&Ks[16 + l15][c * 32], h16);
      s0 = WMMA_BF16(qf[c], bk0, s0);
      s1 = WMMA_BF16(qf[c], bk1, s1);
    }

    // Online softmax over the 32 new keys (band + bounds + padding mask).
    const int j0 = kb + l15, j1 = j0 + 16;
    const bool in0 = (j0 >= 0) && (j0 < 4096);
    const bool in1 = (j1 >= 0) && (j1 < 4096);
    const bool pm0 = in0 && (pm[in0 ? j0 : 0] != 0);
    const bool pm1 = in1 && (pm[in1 ? j1 : 0] != 0);
#pragma unroll
    for (int i = 0; i < 8; ++i) {
      int qi = q0 + wv * 16 + h8 + i;
      int d0 = j0 - qi, d1 = j1 - qi;
      bool ok0 = pm0 && (d0 >= -256) && (d0 <= 256);
      bool ok1 = pm1 && (d1 >= -256) && (d1 <= 256);
      float x0 = ok0 ? s0[i] : -1e30f;
      float x1 = ok1 ? s1[i] : -1e30f;
      float tm = fmaxf(x0, x1);
#pragma unroll
      for (int off = 8; off > 0; off >>= 1)
        tm = fmaxf(tm, __shfl_xor(tm, off, 32));  // reduce across 16-lane half
      float mn   = fmaxf(mrun[i], tm);
      float corr = __expf(mrun[i] - mn);
      float e0 = ok0 ? __expf(x0 - mn) : 0.f;
      float e1 = ok1 ? __expf(x1 - mn) : 0.f;
      float rs = e0 + e1;
#pragma unroll
      for (int off = 8; off > 0; off >>= 1) rs += __shfl_xor(rs, off, 32);
      lrun[i] = lrun[i] * corr + rs;
      mrun[i] = mn;
#pragma unroll
      for (int t = 0; t < 4; ++t) o[t][i] *= corr;
      // C-layout -> row-major P in this wave's LDS slot (wave-in-order LDS).
      Ps[wv][h8 + i][l15]      = (__bf16)e0;
      Ps[wv][h8 + i][16 + l15] = (__bf16)e1;
    }

    // ctx += P(16x32) @ V(32x64): one A-frag of P, four d-tiles of V.
    v16bf pa = load_a_frag(&Ps[wv][l15][0], h8);
#pragma unroll
    for (int t = 0; t < 4; ++t) {
      v16bf bvv = load_b_frag(&Vt[t * 16 + l15][0], h16);
      o[t] = WMMA_BF16(pa, bvv, o[t]);
    }
    __syncthreads();  // protect Ks/Vt before next staging
  }

  // Normalize and write ctx in [B,S,E] layout (row = b*S+s, col = head*64+d).
#pragma unroll
  for (int i = 0; i < 8; ++i) {
    float inv = (lrun[i] > 0.f) ? (1.0f / lrun[i]) : 0.f;
    int srow = q0 + wv * 16 + h8 + i;
    size_t rowoff = ((size_t)(bb * 4096 + srow)) * 1024 + head * 64;
#pragma unroll
    for (int t = 0; t < 4; ++t)
      ctx[rowoff + t * 16 + l15] = o[t][i] * inv;
  }
}

// ---------------------------------------------------------------------------
// Kernel 3: out = ctx @ Wo^T + bo.  M=8192, N=1024, K=1024.
// ---------------------------------------------------------------------------
__global__ __launch_bounds__(256) void k_out_gemm(
    const float* __restrict__ A, const float* __restrict__ W,
    const float* __restrict__ bias, float* __restrict__ out) {
  __shared__ __bf16 As[128][40];
  __shared__ __bf16 Bs[128][40];

  const int tid  = threadIdx.x;
  const int lane = tid & 31, wid = tid >> 5;
  const int l15 = lane & 15, hf = lane >> 4;
  const int h8 = hf << 3, h16 = hf << 4;
  const int mbase = blockIdx.y * 128;
  const int nbase = blockIdx.x * 128;
  const int wr = wid >> 1, wc = wid & 1;

  v8f acc[2][4];
#pragma unroll
  for (int mt = 0; mt < 2; ++mt)
#pragma unroll
    for (int nt = 0; nt < 4; ++nt) acc[mt][nt] = zerov8();

  const int r     = tid >> 1;
  const int halfk = (tid & 1) << 4;
  const float* aptr = A + (size_t)(mbase + r) * 1024 + halfk;
  const float* bptr = W + (size_t)(nbase + r) * 1024 + halfk;

  for (int kc = 0; kc < 1024; kc += 32) {
    if (kc + 32 < 1024) {
      __builtin_prefetch(aptr + kc + 32, 0, 1);
      __builtin_prefetch(bptr + kc + 32, 0, 1);
    }
    const float4* a4 = (const float4*)(aptr + kc);
    const float4* b4 = (const float4*)(bptr + kc);
#pragma unroll
    for (int u = 0; u < 4; ++u) {
      float4 fa = a4[u];
      float4 fb = b4[u];
      As[r][halfk + u * 4 + 0] = (__bf16)fa.x;
      As[r][halfk + u * 4 + 1] = (__bf16)fa.y;
      As[r][halfk + u * 4 + 2] = (__bf16)fa.z;
      As[r][halfk + u * 4 + 3] = (__bf16)fa.w;
      Bs[r][halfk + u * 4 + 0] = (__bf16)fb.x;
      Bs[r][halfk + u * 4 + 1] = (__bf16)fb.y;
      Bs[r][halfk + u * 4 + 2] = (__bf16)fb.z;
      Bs[r][halfk + u * 4 + 3] = (__bf16)fb.w;
    }
    __syncthreads();

    v16bf af[2], bv[4];
#pragma unroll
    for (int mt = 0; mt < 2; ++mt)
      af[mt] = load_a_frag(&As[wr * 32 + mt * 16 + l15][0], h8);
#pragma unroll
    for (int nt = 0; nt < 4; ++nt)
      bv[nt] = load_b_frag(&Bs[wc * 64 + nt * 16 + l15][0], h16);
#pragma unroll
    for (int mt = 0; mt < 2; ++mt)
#pragma unroll
      for (int nt = 0; nt < 4; ++nt)
        acc[mt][nt] = WMMA_BF16(af[mt], bv[nt], acc[mt][nt]);
    __syncthreads();
  }

#pragma unroll
  for (int mt = 0; mt < 2; ++mt)
#pragma unroll
    for (int nt = 0; nt < 4; ++nt) {
      int nglob = nbase + wc * 64 + nt * 16 + l15;
      float bval = bias[nglob];
#pragma unroll
      for (int i = 0; i < 8; ++i) {
        int mrow = mbase + wr * 32 + mt * 16 + h8 + i;
        out[(size_t)mrow * 1024 + nglob] = acc[mt][nt][i] + bval;
      }
    }
}

// ---------------------------------------------------------------------------
// Host launcher. Inputs: x, padding_mask, Wqkv, bqkv, Wo, bo (fp32/int32).
// Workspace: q,k,v [B,H,S,64] + ctx [B,S,E] = 4 * 8,388,608 floats (128 MB).
// ---------------------------------------------------------------------------
extern "C" void kernel_launch(void* const* d_in, const int* in_sizes, int n_in,
                              void* d_out, int out_size, void* d_ws, size_t ws_size,
                              hipStream_t stream) {
  const float* x     = (const float*)d_in[0];
  const int*   pmask = (const int*)d_in[1];
  const float* Wqkv  = (const float*)d_in[2];
  const float* bqkv  = (const float*)d_in[3];
  const float* Wo    = (const float*)d_in[4];
  const float* bo    = (const float*)d_in[5];
  float* out = (float*)d_out;
  float* ws  = (float*)d_ws;

  const size_t per = (size_t)2 * 16 * 4096 * 64;  // 8,388,608 floats
  float* qws = ws;
  float* kws = ws + per;
  float* vws = ws + 2 * per;
  float* cws = ws + 3 * per;

  // qkv projection: M=8192, N=3072
  k_qkv_gemm<<<dim3(24, 64), 256, 0, stream>>>(x, Wqkv, bqkv, qws, kws, vws);
  // banded attention: (S/128, H, B)
  k_attn<<<dim3(32, 16, 2), 256, 0, stream>>>(qws, kws, vws, pmask, cws);
  // output projection: M=8192, N=1024
  k_out_gemm<<<dim3(8, 64), 256, 0, stream>>>(cws, Wo, bo, out);
}
